// Net_73641509257272
// MI455X (gfx1250) — matrix operand order repaired and checked by reading, not compiled, not run on hardware
//
#include <hip/hip_runtime.h>
#include <hip/hip_bf16.h>

// ---------------------------------------------------------------------------
// Leaky RNN: U = x@Wi^T + bi (GEMM), 512-step recurrent scan with per-step
// GEMM h@Wh^T fused with state update + ReLU, then Y = act@Wo^T + bo (GEMM).
// All matmuls use v_wmma_f32_16x16x32_bf16 (wave32), 2-deep software-pipelined
// K loops. Large GEMMs use 32x64 wave tiles (1.5 b128 loads per WMMA); the
// latency-critical recurrent step uses 16x64 tiles for max block parallelism.
// ---------------------------------------------------------------------------

typedef __attribute__((ext_vector_type(16))) __bf16 v16bf;
typedef __attribute__((ext_vector_type(8)))  __bf16 v8bf;
typedef __attribute__((ext_vector_type(8)))  float  v8f;
typedef __attribute__((ext_vector_type(4)))  float  fx4;

#define T_ 512
#define B_ 256
#define I_ 256
#define H_ 1024
#define O_ 128
#define ALPHA_ 0.2f
#define OMA_   0.8f

// ---- workspace layout (bytes) ----
constexpr size_t U_ELEMS  = (size_t)T_ * B_ * H_;            // 134,217,728 f32
constexpr size_t OFF_U    = 0;
constexpr size_t OFF_WHB  = OFF_U   + U_ELEMS * 4;           // Wh bf16 [H,H]
constexpr size_t OFF_WIB  = OFF_WHB + (size_t)H_ * H_ * 2;   // Wi bf16 [H,I]
constexpr size_t OFF_WOB  = OFF_WIB + (size_t)H_ * I_ * 2;   // Wo bf16 [O,H]
constexpr size_t OFF_XB   = OFF_WOB + (size_t)O_ * H_ * 2;   // x  bf16 [T*B,I]
constexpr size_t OFF_ST   = OFF_XB  + (size_t)T_ * B_ * I_ * 2; // state f32 [B,H]
constexpr size_t OFF_HP   = OFF_ST  + (size_t)B_ * H_ * 4;   // hidden bf16 x2 [B,H]

// ---------------------------------------------------------------------------
// Fragment loaders matching the CDNA5 WMMA VGPR layouts (ISA 7.12.2).
// A (16x32 bf16, M=lane&15):   half=0 -> K {kc..kc+7, kc+16..kc+23}
//                              half=1 -> K {kc+8..kc+15, kc+24..kc+31}
// B (32x16 bf16, N=lane&15):   half=0 -> K {kc..kc+7, kc+16..kc+23}, half=1 -> K {kc+8..kc+15, kc+24..kc+31}
// Sources are [rows, K] row-major so every span is contiguous.
// ---------------------------------------------------------------------------
__device__ __forceinline__ v16bf load_a_bf16(const __bf16* __restrict__ rowp,
                                             int kc, int half) {
  v8bf a0 = *(const v8bf*)(rowp + kc + half * 8);
  v8bf a1 = *(const v8bf*)(rowp + kc + 16 + half * 8);
  v16bf r;
#pragma unroll
  for (int e = 0; e < 8; ++e) { r[e] = a0[e]; r[8 + e] = a1[e]; }
  return r;
}

__device__ __forceinline__ v16bf load_a_f32(const float* __restrict__ rowp,
                                            int kc, int half) {
  const fx4* p0 = (const fx4*)(rowp + kc + half * 8);
  const fx4* p1 = (const fx4*)(rowp + kc + 16 + half * 8);
  fx4 f0 = p0[0], f1 = p0[1], f2 = p1[0], f3 = p1[1];
  v16bf r;
#pragma unroll
  for (int e = 0; e < 4; ++e) {
    r[e]      = (__bf16)f0[e];
    r[4 + e]  = (__bf16)f1[e];
    r[8 + e]  = (__bf16)f2[e];
    r[12 + e] = (__bf16)f3[e];
  }
  return r;
}

__device__ __forceinline__ v16bf load_b_bf16(const __bf16* __restrict__ wrow,
                                             int kc, int half) {
  v8bf b0 = *(const v8bf*)(wrow + kc + half * 8);  // same K split as A
  v8bf b1 = *(const v8bf*)(wrow + kc + 16 + half * 8);
  v16bf r;
#pragma unroll
  for (int e = 0; e < 8; ++e) { r[e] = b0[e]; r[8 + e] = b1[e]; }
  return r;
}

#define WMMA_BF16(A, Bm, C) \
  __builtin_amdgcn_wmma_f32_16x16x32_bf16(false, (A), false, (Bm), (short)0, (C), false, false)

// ---------------------------------------------------------------------------
// 16x64 wave GEMM core (for the latency-critical recurrent step).
// 2-deep pipeline: loads for chunk k+32/k+64 in flight during chunk-k WMMAs.
// ---------------------------------------------------------------------------
template <int K>
__device__ __forceinline__ void wave_gemm16(
    const __bf16* __restrict__ arow,
    const __bf16* __restrict__ w0, const __bf16* __restrict__ w1,
    const __bf16* __restrict__ w2, const __bf16* __restrict__ w3,
    int half, v8f (&acc)[4]) {
  v16bf aA  = load_a_bf16(arow, 0, half);
  v16bf bA0 = load_b_bf16(w0, 0, half), bA1 = load_b_bf16(w1, 0, half);
  v16bf bA2 = load_b_bf16(w2, 0, half), bA3 = load_b_bf16(w3, 0, half);
  v16bf aB, bB0, bB1, bB2, bB3;

  int kc = 0;
#pragma unroll 1
  for (; kc < K - 64; kc += 64) {
    aB  = load_a_bf16(arow, kc + 32, half);
    bB0 = load_b_bf16(w0, kc + 32, half); bB1 = load_b_bf16(w1, kc + 32, half);
    bB2 = load_b_bf16(w2, kc + 32, half); bB3 = load_b_bf16(w3, kc + 32, half);
    acc[0] = WMMA_BF16(aA, bA0, acc[0]); acc[1] = WMMA_BF16(aA, bA1, acc[1]);
    acc[2] = WMMA_BF16(aA, bA2, acc[2]); acc[3] = WMMA_BF16(aA, bA3, acc[3]);
    aA  = load_a_bf16(arow, kc + 64, half);
    bA0 = load_b_bf16(w0, kc + 64, half); bA1 = load_b_bf16(w1, kc + 64, half);
    bA2 = load_b_bf16(w2, kc + 64, half); bA3 = load_b_bf16(w3, kc + 64, half);
    acc[0] = WMMA_BF16(aB, bB0, acc[0]); acc[1] = WMMA_BF16(aB, bB1, acc[1]);
    acc[2] = WMMA_BF16(aB, bB2, acc[2]); acc[3] = WMMA_BF16(aB, bB3, acc[3]);
  }
  aB  = load_a_bf16(arow, K - 32, half);
  bB0 = load_b_bf16(w0, K - 32, half); bB1 = load_b_bf16(w1, K - 32, half);
  bB2 = load_b_bf16(w2, K - 32, half); bB3 = load_b_bf16(w3, K - 32, half);
  acc[0] = WMMA_BF16(aA, bA0, acc[0]); acc[1] = WMMA_BF16(aA, bA1, acc[1]);
  acc[2] = WMMA_BF16(aA, bA2, acc[2]); acc[3] = WMMA_BF16(aA, bA3, acc[3]);
  acc[0] = WMMA_BF16(aB, bB0, acc[0]); acc[1] = WMMA_BF16(aB, bB1, acc[1]);
  acc[2] = WMMA_BF16(aB, bB2, acc[2]); acc[3] = WMMA_BF16(aB, bB3, acc[3]);
}

// ---------------------------------------------------------------------------
// 32x64 wave GEMM core (high arithmetic intensity for the big GEMMs).
// acc[0..3] = rows rowBase..+15 ; acc[4..7] = rows rowBase+16..+31.
// Each B fragment feeds two WMMAs -> 16 WMMAs per 12 fragment loads.
// ---------------------------------------------------------------------------
template <int K, bool AF32>
__device__ __forceinline__ void wave_gemm32(
    const void* __restrict__ arow0, const void* __restrict__ arow1,
    const __bf16* __restrict__ w0, const __bf16* __restrict__ w1,
    const __bf16* __restrict__ w2, const __bf16* __restrict__ w3,
    int half, v8f (&acc)[8]) {
  auto la = [&](const void* p, int kc) -> v16bf {
    if constexpr (AF32) return load_a_f32((const float*)p, kc, half);
    else                return load_a_bf16((const __bf16*)p, kc, half);
  };
  v16bf aA0 = la(arow0, 0), aA1 = la(arow1, 0);
  v16bf bA0 = load_b_bf16(w0, 0, half), bA1 = load_b_bf16(w1, 0, half);
  v16bf bA2 = load_b_bf16(w2, 0, half), bA3 = load_b_bf16(w3, 0, half);
  v16bf aB0, aB1, bB0, bB1, bB2, bB3;

  int kc = 0;
#pragma unroll 1
  for (; kc < K - 64; kc += 64) {
    aB0 = la(arow0, kc + 32); aB1 = la(arow1, kc + 32);
    bB0 = load_b_bf16(w0, kc + 32, half); bB1 = load_b_bf16(w1, kc + 32, half);
    bB2 = load_b_bf16(w2, kc + 32, half); bB3 = load_b_bf16(w3, kc + 32, half);
    acc[0] = WMMA_BF16(aA0, bA0, acc[0]); acc[4] = WMMA_BF16(aA1, bA0, acc[4]);
    acc[1] = WMMA_BF16(aA0, bA1, acc[1]); acc[5] = WMMA_BF16(aA1, bA1, acc[5]);
    acc[2] = WMMA_BF16(aA0, bA2, acc[2]); acc[6] = WMMA_BF16(aA1, bA2, acc[6]);
    acc[3] = WMMA_BF16(aA0, bA3, acc[3]); acc[7] = WMMA_BF16(aA1, bA3, acc[7]);
    aA0 = la(arow0, kc + 64); aA1 = la(arow1, kc + 64);
    bA0 = load_b_bf16(w0, kc + 64, half); bA1 = load_b_bf16(w1, kc + 64, half);
    bA2 = load_b_bf16(w2, kc + 64, half); bA3 = load_b_bf16(w3, kc + 64, half);
    acc[0] = WMMA_BF16(aB0, bB0, acc[0]); acc[4] = WMMA_BF16(aB1, bB0, acc[4]);
    acc[1] = WMMA_BF16(aB0, bB1, acc[1]); acc[5] = WMMA_BF16(aB1, bB1, acc[5]);
    acc[2] = WMMA_BF16(aB0, bB2, acc[2]); acc[6] = WMMA_BF16(aB1, bB2, acc[6]);
    acc[3] = WMMA_BF16(aB0, bB3, acc[3]); acc[7] = WMMA_BF16(aB1, bB3, acc[7]);
  }
  aB0 = la(arow0, K - 32); aB1 = la(arow1, K - 32);
  bB0 = load_b_bf16(w0, K - 32, half); bB1 = load_b_bf16(w1, K - 32, half);
  bB2 = load_b_bf16(w2, K - 32, half); bB3 = load_b_bf16(w3, K - 32, half);
  acc[0] = WMMA_BF16(aA0, bA0, acc[0]); acc[4] = WMMA_BF16(aA1, bA0, acc[4]);
  acc[1] = WMMA_BF16(aA0, bA1, acc[1]); acc[5] = WMMA_BF16(aA1, bA1, acc[5]);
  acc[2] = WMMA_BF16(aA0, bA2, acc[2]); acc[6] = WMMA_BF16(aA1, bA2, acc[6]);
  acc[3] = WMMA_BF16(aA0, bA3, acc[3]); acc[7] = WMMA_BF16(aA1, bA3, acc[7]);
  acc[0] = WMMA_BF16(aB0, bB0, acc[0]); acc[4] = WMMA_BF16(aB1, bB0, acc[4]);
  acc[1] = WMMA_BF16(aB0, bB1, acc[1]); acc[5] = WMMA_BF16(aB1, bB1, acc[5]);
  acc[2] = WMMA_BF16(aB0, bB2, acc[2]); acc[6] = WMMA_BF16(aB1, bB2, acc[6]);
  acc[3] = WMMA_BF16(aB0, bB3, acc[3]); acc[7] = WMMA_BF16(aB1, bB3, acc[7]);
}

// ---------------------------------------------------------------------------
// Prep: bf16-convert x/Wi/Wh/Wo, zero state + ping buffer (deterministic).
// ---------------------------------------------------------------------------
__global__ void k_prep(const float* __restrict__ x,  const float* __restrict__ Wi,
                       const float* __restrict__ Wh, const float* __restrict__ Wo,
                       __bf16* __restrict__ xb,  __bf16* __restrict__ wib,
                       __bf16* __restrict__ whb, __bf16* __restrict__ wob,
                       float* __restrict__ state, __bf16* __restrict__ hping) {
  const long stride = (long)gridDim.x * blockDim.x;
  const long tid = (long)blockIdx.x * blockDim.x + threadIdx.x;
  for (long i = tid; i < (long)T_ * B_ * I_; i += stride) xb[i]  = (__bf16)x[i];
  for (long i = tid; i < (long)H_ * I_;      i += stride) wib[i] = (__bf16)Wi[i];
  for (long i = tid; i < (long)H_ * H_;      i += stride) whb[i] = (__bf16)Wh[i];
  for (long i = tid; i < (long)O_ * H_;      i += stride) wob[i] = (__bf16)Wo[i];
  for (long i = tid; i < (long)B_ * H_;      i += stride) {
    state[i] = 0.0f; hping[i] = (__bf16)0.0f;
  }
}

// ---------------------------------------------------------------------------
// Phase 1: U[m,h] = x[m,:].Wi[h,:] + bi[h];  M=131072 N=1024 K=256
// wave tile 32x64, block = 8 waves -> 256x64 tile; grid (16, 512).
// ---------------------------------------------------------------------------
__global__ __launch_bounds__(256) void k_inproj(
    const __bf16* __restrict__ xb, const __bf16* __restrict__ wib,
    const float* __restrict__ bi, float* __restrict__ U) {
  const int lane = threadIdx.x & 31, wave = threadIdx.x >> 5;
  const int half = lane >> 4, l16 = lane & 15;
  const long rowBase = (long)blockIdx.y * 256 + wave * 32;
  const int  colBase = blockIdx.x * 64;

  v8f acc[8] = {};
  wave_gemm32<I_, false>(xb + (rowBase + l16) * I_,
                         xb + (rowBase + 16 + l16) * I_,
                         wib + (size_t)(colBase +  0 + l16) * I_,
                         wib + (size_t)(colBase + 16 + l16) * I_,
                         wib + (size_t)(colBase + 32 + l16) * I_,
                         wib + (size_t)(colBase + 48 + l16) * I_,
                         half, acc);

  const long mBase = rowBase + half * 8;
#pragma unroll
  for (int j = 0; j < 4; ++j) {
    const int n = colBase + j * 16 + l16;
    const float bn = bi[n];
#pragma unroll
    for (int r = 0; r < 8; ++r) {
      U[(mBase + r) * H_ + n]      = acc[j][r]     + bn;
      U[(mBase + 16 + r) * H_ + n] = acc[4 + j][r] + bn;
    }
  }
}

// ---------------------------------------------------------------------------
// Phase 2 (per timestep t): r = h_prev @ Wh^T ;  tot = U_t + r + bh
// state = 0.8*state + 0.2*tot ; act = relu(state)
// M=256 N=1024 K=1024; 16x64 wave tiles -> 32 blocks for step latency.
// ---------------------------------------------------------------------------
__global__ __launch_bounds__(256) void k_step(
    const __bf16* __restrict__ hprev, const __bf16* __restrict__ whb,
    const float* __restrict__ Ut, const float* __restrict__ bh,
    float* __restrict__ state, float* __restrict__ act_out,
    __bf16* __restrict__ hnext) {
  const int lane = threadIdx.x & 31, wave = threadIdx.x >> 5;
  const int half = lane >> 4, l16 = lane & 15;
  const int rowBase = blockIdx.y * 128 + wave * 16;
  const int colBase = blockIdx.x * 64;
  const __bf16* arow = hprev + (size_t)(rowBase + l16) * H_;

  v8f acc[4] = {};
  wave_gemm16<H_>(arow,
                  whb + (size_t)(colBase +  0 + l16) * H_,
                  whb + (size_t)(colBase + 16 + l16) * H_,
                  whb + (size_t)(colBase + 32 + l16) * H_,
                  whb + (size_t)(colBase + 48 + l16) * H_,
                  half, acc);

  const int mBase = rowBase + half * 8;
#pragma unroll
  for (int j = 0; j < 4; ++j) {
    const int n = colBase + j * 16 + l16;
    const float bn = bh[n];
#pragma unroll
    for (int r = 0; r < 8; ++r) {
      const long idx = (long)(mBase + r) * H_ + n;
      const float tot = acc[j][r] + Ut[idx] + bn;
      const float s   = state[idx] * OMA_ + tot * ALPHA_;
      const float a_  = fmaxf(s, 0.0f);
      state[idx]   = s;
      act_out[idx] = a_;
      hnext[idx]   = (__bf16)a_;
    }
  }
}

// ---------------------------------------------------------------------------
// Phase 3: Y[m,o] = act[m,:].Wo[o,:] + bo[o];  M=131072 N=128 K=1024
// A is fp32 activity (from d_out), converted to bf16 in-register.
// wave tile 32x64; grid (2, 512).
// ---------------------------------------------------------------------------
__global__ __launch_bounds__(256) void k_outproj(
    const float* __restrict__ act, const __bf16* __restrict__ wob,
    const float* __restrict__ bo, float* __restrict__ y) {
  const int lane = threadIdx.x & 31, wave = threadIdx.x >> 5;
  const int half = lane >> 4, l16 = lane & 15;
  const long rowBase = (long)blockIdx.y * 256 + wave * 32;
  const int  colBase = blockIdx.x * 64;

  v8f acc[8] = {};
  wave_gemm32<H_, true>(act + (rowBase + l16) * (size_t)H_,
                        act + (rowBase + 16 + l16) * (size_t)H_,
                        wob + (size_t)(colBase +  0 + l16) * H_,
                        wob + (size_t)(colBase + 16 + l16) * H_,
                        wob + (size_t)(colBase + 32 + l16) * H_,
                        wob + (size_t)(colBase + 48 + l16) * H_,
                        half, acc);

  const long mBase = rowBase + half * 8;
#pragma unroll
  for (int j = 0; j < 4; ++j) {
    const int n = colBase + j * 16 + l16;
    const float bn = bo[n];
#pragma unroll
    for (int r = 0; r < 8; ++r) {
      y[(mBase + r) * O_ + n]      = acc[j][r]     + bn;
      y[(mBase + 16 + r) * O_ + n] = acc[4 + j][r] + bn;
    }
  }
}

// ---------------------------------------------------------------------------
extern "C" void kernel_launch(void* const* d_in, const int* in_sizes, int n_in,
                              void* d_out, int out_size, void* d_ws, size_t ws_size,
                              hipStream_t stream) {
  const float* x  = (const float*)d_in[0];
  const float* Wi = (const float*)d_in[1];
  const float* bi = (const float*)d_in[2];
  const float* Wh = (const float*)d_in[3];
  const float* bh = (const float*)d_in[4];
  const float* Wo = (const float*)d_in[5];
  const float* bo = (const float*)d_in[6];

  char* ws = (char*)d_ws;
  float*  U     = (float*) (ws + OFF_U);
  __bf16* whb   = (__bf16*)(ws + OFF_WHB);
  __bf16* wib   = (__bf16*)(ws + OFF_WIB);
  __bf16* wob   = (__bf16*)(ws + OFF_WOB);
  __bf16* xb    = (__bf16*)(ws + OFF_XB);
  float*  state = (float*) (ws + OFF_ST);
  __bf16* hping = (__bf16*)(ws + OFF_HP);

  float* out = (float*)d_out;                       // [T,B,O]
  float* act = out + (size_t)T_ * B_ * O_;          // [T,B,H] (second output)

  // 0) convert weights/x to bf16, zero state & ping buffer
  k_prep<<<2048, 256, 0, stream>>>(x, Wi, Wh, Wo, xb, wib, whb, wob, state, hping);

  // 1) input projection for all timesteps
  dim3 g1(H_ / 64, (T_ * B_) / 256);                // (16, 512)
  k_inproj<<<g1, 256, 0, stream>>>(xb, wib, bi, U);

  // 2) recurrent scan: one launch per timestep (sequential dependency)
  dim3 g2(H_ / 64, B_ / 128);                       // (16, 2)
  for (int t = 0; t < T_; ++t) {
    const __bf16* hp = hping + (size_t)(t & 1) * B_ * H_;
    __bf16*       hn = hping + (size_t)((t + 1) & 1) * B_ * H_;
    k_step<<<g2, 256, 0, stream>>>(hp, whb, U + (size_t)t * B_ * H_, bh,
                                   state, act + (size_t)t * B_ * H_, hn);
  }

  // 3) output projection over all timesteps
  dim3 g3(O_ / 64, (T_ * B_) / 256);                // (2, 512)
  k_outproj<<<g3, 256, 0, stream>>>(act, wob, bo, out);
}
